// TinyRecursiveModel_13194139533675
// MI455X (gfx1250) — matrix operand order repaired and checked by reading, not compiled, not run on hardware
//
#include <hip/hip_runtime.h>

// ---------------- constants from the reference ----------------
constexpr int BB = 4096;   // batch
constexpr int SS = 64;     // seq
constexpr int INq = 256;   // IN
constexpr int DD = 512;    // D
constexpr int AANS = 128;  // A (answer dim)
constexpr int OS = 4;      // max_outer_steps
constexpr int IS = 8;      // max_inner_steps

typedef __attribute__((ext_vector_type(16))) _Float16 v16h;
typedef __attribute__((ext_vector_type(8)))  _Float16 v8h;
typedef __attribute__((ext_vector_type(8)))  float    v8f;
typedef __attribute__((ext_vector_type(4)))  float    v4f;

__device__ inline v8f zero_v8f() {
    v8f z;
#pragma unroll
    for (int i = 0; i < 8; ++i) z[i] = 0.0f;
    return z;
}

// ---- A-fragment (16x32 f16, M x K): lane = row (lane&15), K-half by lane>>4.
// K offsets per lane: f[0..7] = K {0..7}+8g, f[8..15] = K {16..23}+8g.
__device__ inline v16h load_afrag_f16(const _Float16* A, int m0, int k0, int lda) {
    int lane = threadIdx.x & 31;
    int row  = m0 + (lane & 15);
    int g    = lane >> 4;
    const _Float16* p = A + (size_t)row * lda + k0 + 8 * g;
    v8h lo = *(const v8h*)(p);
    v8h hi = *(const v8h*)(p + 16);
    v16h f;
#pragma unroll
    for (int i = 0; i < 8; ++i) { f[i] = lo[i]; f[i + 8] = hi[i]; }
    return f;
}

// Same fragment but converting from f32 source on the fly (steps matrix).
__device__ inline v16h load_afrag_f32(const float* A, int m0, int k0, int lda) {
    int lane = threadIdx.x & 31;
    int row  = m0 + (lane & 15);
    int g    = lane >> 4;
    const float* p = A + (size_t)row * lda + k0 + 8 * g;
    const v4f* q = (const v4f*)p;
    v4f x0 = q[0], x1 = q[1], x2 = q[4], x3 = q[5];
    v16h f;
#pragma unroll
    for (int i = 0; i < 4; ++i) {
        f[i]      = (_Float16)x0[i];
        f[4 + i]  = (_Float16)x1[i];
        f[8 + i]  = (_Float16)x2[i];
        f[12 + i] = (_Float16)x3[i];
    }
    return f;
}

// ---- B-fragment (32x16 f16, K x N) from transposed weight Bt (K x N row-major):
// lane L holds column n = n0 + (L&15); f[0..7] = K {0..7}+8g, f[8..15] = K {16..23}+8g.
__device__ inline v16h load_bfrag(const _Float16* Bt, int n0, int k0, int ldn) {
    int lane = threadIdx.x & 31;
    int g = lane >> 4;
    const _Float16* p = Bt + (size_t)(k0 + 8 * g) * ldn + n0 + (lane & 15);
    v16h f;
#pragma unroll
    for (int i = 0; i < 8; ++i) { f[i] = p[(size_t)i * ldn]; f[i + 8] = p[(size_t)(16 + i) * ldn]; }
    return f;
}

// ---------------- weight convert + transpose:  W (N x K f32) -> Wt (K x N f16)
__global__ void cvt_transpose_kernel(const float* __restrict__ W,
                                     _Float16* __restrict__ Wt, int Nn, int Kk) {
    int idx = blockIdx.x * 256 + threadIdx.x;
    if (idx < Nn * Kk) {
        int k = idx / Nn, n = idx - k * Nn;
        Wt[idx] = (_Float16)W[(size_t)n * Kk + k];
    }
}

// ---------------- fused steps_emb GEMM + masked-mean + gather (one WG per batch b)
__global__ __launch_bounds__(256) void emb_kernel(
    const float* __restrict__ steps, const _Float16* __restrict__ Wstep_t,
    const float* __restrict__ b_step, const int* __restrict__ lengths,
    float* __restrict__ cmean, _Float16* __restrict__ sel /* [IS][BB][DD] */) {
    int b    = blockIdx.x;
    int wave = threadIdx.x >> 5;
    int lane = threadIdx.x & 31;
    int laneN = lane & 15;
    int g    = lane >> 4;
    int len  = lengths[b];
    if (len < 1) len = 1;
    if (len > SS) len = SS;
    const float* Ab = steps + (size_t)b * SS * INq;
    float csum[4] = {0.f, 0.f, 0.f, 0.f};

    for (int mt = 0; mt < 4; ++mt) {                 // 4 M-tiles cover s = 0..63
        v8f acc[4];
#pragma unroll
        for (int j = 0; j < 4; ++j) acc[j] = zero_v8f();
        // cap pipelining so the kernel stays under 256 VGPRs
#pragma clang loop unroll_count(2)
        for (int kk = 0; kk < INq; kk += 32) {
            v16h a = load_afrag_f32(Ab, mt * 16, kk, INq);
#pragma unroll
            for (int j = 0; j < 4; ++j) {
                int n0 = wave * 64 + j * 16;
                v16h bf = load_bfrag(Wstep_t, n0, kk, DD);
                acc[j] = __builtin_amdgcn_wmma_f32_16x16x32_f16(
                    false, a, false, bf, (short)0, acc[j], false, false);
            }
        }
        // epilogue: bias, masked-mean accumulation, gather of selected rows
#pragma unroll
        for (int j = 0; j < 4; ++j) {
            int n = wave * 64 + j * 16 + laneN;
            float bias = b_step[n];
#pragma unroll
            for (int r = 0; r < 8; ++r) {
                int s = mt * 16 + r + 8 * g;
                float v = acc[j][r] + bias;
                if (s < len) csum[j] += v;
                // selected[ji] = emb row min(ji, len-1)
                if (s < IS && s < len)
                    sel[((size_t)s * BB + b) * DD + n] = (_Float16)v;
                if (s == len - 1) {
                    for (int ji = len; ji < IS; ++ji)
                        sel[((size_t)ji * BB + b) * DD + n] = (_Float16)v;
                }
            }
        }
    }
    float inv = 1.0f / (float)len;
#pragma unroll
    for (int j = 0; j < 4; ++j) {
        float other = __shfl_xor(csum[j], 16, 32);   // combine the two halves
        if (g == 0) {
            int n = wave * 64 + j * 16 + laneN;
            cmean[(size_t)b * DD + n] = (csum[j] + other) * inv;
        }
    }
}

// ---------------- block reduction (256 threads, 8 waves of 32)
__device__ inline void block_reduce2(float& a, float& b) {
    __shared__ float smA[8], smB[8];
#pragma unroll
    for (int off = 16; off; off >>= 1) {
        a += __shfl_xor(a, off, 32);
        b += __shfl_xor(b, off, 32);
    }
    int wave = threadIdx.x >> 5, lane = threadIdx.x & 31;
    __syncthreads();
    if (lane == 0) { smA[wave] = a; smB[wave] = b; }
    __syncthreads();
    float ta = 0.f, tb = 0.f;
#pragma unroll
    for (int i = 0; i < 8; ++i) { ta += smA[i]; tb += smB[i]; }
    a = ta; b = tb;
}

// ---------------- z = LN(start*W_start + b_start + cmean, g_init, bn_init)
__global__ __launch_bounds__(256) void zinit_kernel(
    const float* __restrict__ start, const float* __restrict__ W_start,
    const float* __restrict__ b_start, const float* __restrict__ cmean,
    const float* __restrict__ g_init, const float* __restrict__ bn_init,
    float* __restrict__ z) {
    int b = blockIdx.x, tid = threadIdx.x;
    float s0 = start[b];
    int d0 = tid, d1 = tid + 256;
    float v0 = s0 * W_start[d0] + b_start[d0] + cmean[(size_t)b * DD + d0];
    float v1 = s0 * W_start[d1] + b_start[d1] + cmean[(size_t)b * DD + d1];
    float s = v0 + v1, q = v0 * v0 + v1 * v1;
    block_reduce2(s, q);
    float mean = s * (1.f / DD);
    float var  = q * (1.f / DD) - mean * mean;
    float rs   = rsqrtf(var + 1e-5f);
    z[(size_t)b * DD + d0] = (v0 - mean) * rs * g_init[d0] + bn_init[d0];
    z[(size_t)b * DD + d1] = (v1 - mean) * rs * g_init[d1] + bn_init[d1];
}

// ---------------- generic WMMA GEMM:  C = act(A @ Bt + bias)
// A: M x K f16 row-major. Bt: K x N f16 row-major (pre-transposed weight).
// Each wave computes a 32(M) x 64(N) block: 2 M-tiles x 4 N-tiles, so the 4
// B-fragments are reused across 2 A-fragments (8 WMMAs per 6 fragment loads).
// WG tile 128(M) x 128(N); grid = (M/128, N/128). act: 0=none, 1=silu.
__global__ __launch_bounds__(256) void gemm_wmma(
    const _Float16* __restrict__ Ah, int lda,
    const _Float16* __restrict__ Bt, int ldb,
    const float* __restrict__ bias,
    _Float16* __restrict__ C16, float* __restrict__ C32, int ldc,
    int K, int act) {
    int wave = threadIdx.x >> 5, lane = threadIdx.x & 31;
    int laneN = lane & 15, g = lane >> 4;
    int nbase = blockIdx.y * 128 + (wave & 1) * 64;
    int m0 = blockIdx.x * 128 + (wave >> 1) * 32;

    v8f acc[2][4];
#pragma unroll
    for (int t = 0; t < 2; ++t)
#pragma unroll
        for (int j = 0; j < 4; ++j) acc[t][j] = zero_v8f();

    for (int k0 = 0; k0 < K; k0 += 32) {
        if (k0 + 32 < K)   // hint the next A K-slab into cache (global_prefetch)
            __builtin_prefetch(Ah + (size_t)(m0 + (threadIdx.x & 31)) * lda + k0 + 32, 0, 0);
        v16h a0 = load_afrag_f16(Ah, m0, k0, lda);
        v16h a1 = load_afrag_f16(Ah, m0 + 16, k0, lda);
#pragma unroll
        for (int j = 0; j < 4; ++j) {
            v16h bf = load_bfrag(Bt, nbase + j * 16, k0, ldb);
            acc[0][j] = __builtin_amdgcn_wmma_f32_16x16x32_f16(
                false, a0, false, bf, (short)0, acc[0][j], false, false);
            acc[1][j] = __builtin_amdgcn_wmma_f32_16x16x32_f16(
                false, a1, false, bf, (short)0, acc[1][j], false, false);
        }
    }
#pragma unroll
    for (int t = 0; t < 2; ++t) {
#pragma unroll
        for (int j = 0; j < 4; ++j) {
            int n = nbase + j * 16 + laneN;
            float bv = bias ? bias[n] : 0.f;
#pragma unroll
            for (int r = 0; r < 8; ++r) {
                int m = m0 + t * 16 + r + 8 * g;
                float v = acc[t][j][r] + bv;
                if (act == 1) v = v / (1.f + __expf(-v));
                if (C16) C16[(size_t)m * ldc + n] = (_Float16)v;
                if (C32) C32[(size_t)m * ldc + n] = v;
            }
        }
    }
}

// ---------------- inner loop for one outer step: 8x { z = LN(z + update[j]) }
// update depends only on `selected`, so it was precomputed once by the GEMM chain.
__global__ __launch_bounds__(256) void inner_kernel(
    const float* __restrict__ update /* [IS][BB][DD] */,
    const float* __restrict__ g_inner, const float* __restrict__ bn_inner,
    float* __restrict__ z, _Float16* __restrict__ z16,
    float* __restrict__ latents, int o) {
    int b = blockIdx.x, tid = threadIdx.x;
    int d0 = tid, d1 = tid + 256;
    float v0 = z[(size_t)b * DD + d0];
    float v1 = z[(size_t)b * DD + d1];
    float gi0 = g_inner[d0], gi1 = g_inner[d1];
    float bi0 = bn_inner[d0], bi1 = bn_inner[d1];
    for (int j = 0; j < IS; ++j) {
        float a0 = v0 + update[((size_t)j * BB + b) * DD + d0];
        float a1 = v1 + update[((size_t)j * BB + b) * DD + d1];
        float s = a0 + a1, q = a0 * a0 + a1 * a1;
        block_reduce2(s, q);
        float mean = s * (1.f / DD);
        float var  = q * (1.f / DD) - mean * mean;
        float rs   = rsqrtf(var + 1e-5f);
        v0 = (a0 - mean) * rs * gi0 + bi0;
        v1 = (a1 - mean) * rs * gi1 + bi1;
        float* lat = latents + ((size_t)b * (OS * IS) + o * IS + j) * DD;
        lat[d0] = v0;
        lat[d1] = v1;
    }
    z[(size_t)b * DD + d0] = v0;
    z[(size_t)b * DD + d1] = v1;
    z16[(size_t)b * DD + d0] = (_Float16)v0;
    z16[(size_t)b * DD + d1] = (_Float16)v1;
}

// ---------------- outer epilogue: z = LN(z+u, g_outer); ln_ans = LN(z, g_ans);
// halt[b,o] = LN(z, g_halt) . W_halt + b_halt
__global__ __launch_bounds__(256) void outerz_kernel(
    const float* __restrict__ u,
    const float* __restrict__ g_outer, const float* __restrict__ bn_outer,
    const float* __restrict__ g_ans,   const float* __restrict__ bn_ans,
    const float* __restrict__ g_halt,  const float* __restrict__ bn_halt,
    const float* __restrict__ W_halt,  const float* __restrict__ b_halt,
    float* __restrict__ z, _Float16* __restrict__ z16,
    _Float16* __restrict__ lnans, float* __restrict__ halt_out, int o) {
    int b = blockIdx.x, tid = threadIdx.x;
    int d0 = tid, d1 = tid + 256;
    float a0 = z[(size_t)b * DD + d0] + u[(size_t)b * DD + d0];
    float a1 = z[(size_t)b * DD + d1] + u[(size_t)b * DD + d1];
    float s = a0 + a1, q = a0 * a0 + a1 * a1;
    block_reduce2(s, q);
    float mean = s * (1.f / DD);
    float var  = q * (1.f / DD) - mean * mean;
    float rs   = rsqrtf(var + 1e-5f);
    float zn0 = (a0 - mean) * rs * g_outer[d0] + bn_outer[d0];
    float zn1 = (a1 - mean) * rs * g_outer[d1] + bn_outer[d1];
    z[(size_t)b * DD + d0] = zn0;
    z[(size_t)b * DD + d1] = zn1;
    z16[(size_t)b * DD + d0] = (_Float16)zn0;
    z16[(size_t)b * DD + d1] = (_Float16)zn1;
    // second LN stats over the fresh z
    float s2 = zn0 + zn1, q2 = zn0 * zn0 + zn1 * zn1;
    block_reduce2(s2, q2);
    float m2  = s2 * (1.f / DD);
    float v2  = q2 * (1.f / DD) - m2 * m2;
    float rs2 = rsqrtf(v2 + 1e-5f);
    float n0 = (zn0 - m2) * rs2, n1 = (zn1 - m2) * rs2;
    lnans[(size_t)b * DD + d0] = (_Float16)(n0 * g_ans[d0] + bn_ans[d0]);
    lnans[(size_t)b * DD + d1] = (_Float16)(n1 * g_ans[d1] + bn_ans[d1]);
    float h0 = n0 * g_halt[d0] + bn_halt[d0];
    float h1 = n1 * g_halt[d1] + bn_halt[d1];
    float dot = h0 * W_halt[d0] + h1 * W_halt[d1], dummy = 0.f;
    block_reduce2(dot, dummy);
    if (tid == 0) halt_out[(size_t)b * OS + o] = dot + b_halt[0];
}

// ================================================================ host side
extern "C" void kernel_launch(void* const* d_in, const int* in_sizes, int n_in,
                              void* d_out, int out_size, void* d_ws, size_t ws_size,
                              hipStream_t stream) {
    (void)in_sizes; (void)n_in; (void)out_size; (void)ws_size;
    const float* start   = (const float*)d_in[0];
    const float* steps   = (const float*)d_in[1];
    const int*   lengths = (const int*)  d_in[2];
    // d_in[3], d_in[4]: max_outer/inner steps == OS/IS (module constants)
    const float* W_start = (const float*)d_in[5];
    const float* b_start = (const float*)d_in[6];
    const float* W_step  = (const float*)d_in[7];
    const float* b_step  = (const float*)d_in[8];
    const float* g_init  = (const float*)d_in[9];
    const float* bn_init = (const float*)d_in[10];
    const float* W_val   = (const float*)d_in[11];
    const float* b_val   = (const float*)d_in[12];
    const float* W_vin   = (const float*)d_in[13];
    const float* b_vin   = (const float*)d_in[14];
    const float* W_ao    = (const float*)d_in[15];
    const float* b_ao    = (const float*)d_in[16];
    const float* W_out   = (const float*)d_in[17];
    const float* b_out   = (const float*)d_in[18];
    const float* g_inner = (const float*)d_in[19];
    const float* bn_inner= (const float*)d_in[20];
    const float* W_o1    = (const float*)d_in[21];
    const float* b_o1    = (const float*)d_in[22];
    const float* W_o2    = (const float*)d_in[23];
    const float* b_o2    = (const float*)d_in[24];
    const float* g_outer = (const float*)d_in[25];
    const float* bn_outer= (const float*)d_in[26];
    const float* g_ans   = (const float*)d_in[27];
    const float* bn_ans  = (const float*)d_in[28];
    const float* W_ans   = (const float*)d_in[29];
    const float* b_ans   = (const float*)d_in[30];
    const float* g_halt  = (const float*)d_in[31];
    const float* bn_halt = (const float*)d_in[32];
    const float* W_halt  = (const float*)d_in[33];
    const float* b_halt  = (const float*)d_in[34];

    float* out     = (float*)d_out;
    float* logits  = out;                                  // [B][OS][A]
    float* halt    = out + (size_t)BB * OS * AANS;         // [B][OS]
    float* latents = halt + (size_t)BB * OS;               // [B][OS*IS][D]

    // -------- workspace carve-out (256B aligned slabs)
    char* wsp = (char*)d_ws;
    size_t off = 0;
    auto take = [&](size_t bytes) -> char* {
        char* p = wsp + off;
        off += (bytes + 255) & ~(size_t)255;
        return p;
    };
    _Float16* Wstep_t = (_Float16*)take((size_t)INq * DD * 2);
    _Float16* Wval_t  = (_Float16*)take((size_t)DD * DD * 2);
    _Float16* Wvin_t  = (_Float16*)take((size_t)DD * DD * 2);
    _Float16* Wao_t   = (_Float16*)take((size_t)DD * DD * 2);
    _Float16* Wout_t  = (_Float16*)take((size_t)DD * DD * 2);
    _Float16* Wo1_t   = (_Float16*)take((size_t)DD * DD * 2);
    _Float16* Wo2_t   = (_Float16*)take((size_t)DD * DD * 2);
    _Float16* Wans_t  = (_Float16*)take((size_t)DD * AANS * 2);
    _Float16* sel16   = (_Float16*)take((size_t)IS * BB * DD * 2);
    _Float16* tA      = (_Float16*)take((size_t)IS * BB * DD * 2);
    _Float16* tB      = (_Float16*)take((size_t)IS * BB * DD * 2);
    float*    update  = (float*)   take((size_t)IS * BB * DD * 4);
    float*    cmean   = (float*)   take((size_t)BB * DD * 4);
    float*    z       = (float*)   take((size_t)BB * DD * 4);
    _Float16* z16     = (_Float16*)take((size_t)BB * DD * 2);
    _Float16* h16     = (_Float16*)take((size_t)BB * DD * 2);
    float*    u       = (float*)   take((size_t)BB * DD * 4);
    _Float16* lnans   = (_Float16*)take((size_t)BB * DD * 2);

    // -------- convert + transpose weights to f16 (K x N)
    auto cvt = [&](const float* src, _Float16* dst, int Nn, int Kk) {
        int total = Nn * Kk;
        cvt_transpose_kernel<<<(total + 255) / 256, 256, 0, stream>>>(src, dst, Nn, Kk);
    };
    cvt(W_step, Wstep_t, DD, INq);
    cvt(W_val,  Wval_t,  DD, DD);
    cvt(W_vin,  Wvin_t,  DD, DD);
    cvt(W_ao,   Wao_t,   DD, DD);
    cvt(W_out,  Wout_t,  DD, DD);
    cvt(W_o1,   Wo1_t,   DD, DD);
    cvt(W_o2,   Wo2_t,   DD, DD);
    cvt(W_ans,  Wans_t,  AANS, DD);

    // -------- steps_emb GEMM fused with masked mean + gather of selected rows
    emb_kernel<<<BB, 256, 0, stream>>>(steps, Wstep_t, b_step, lengths, cmean, sel16);
    zinit_kernel<<<BB, 256, 0, stream>>>(start, W_start, b_start, cmean, g_init, bn_init, z);

    // -------- precompute update = (((sel@Wv)+b)@Wvin ... )@Wout  (z-independent!)
    dim3 gch(IS * BB / 128, DD / 128);
    gemm_wmma<<<gch, 256, 0, stream>>>(sel16, DD, Wval_t, DD, b_val, tA, nullptr, DD, DD, 0);
    gemm_wmma<<<gch, 256, 0, stream>>>(tA,    DD, Wvin_t, DD, b_vin, tB, nullptr, DD, DD, 0);
    gemm_wmma<<<gch, 256, 0, stream>>>(tB,    DD, Wao_t,  DD, b_ao,  tA, nullptr, DD, DD, 0);
    gemm_wmma<<<gch, 256, 0, stream>>>(tA,    DD, Wout_t, DD, b_out, nullptr, update, DD, DD, 0);

    // -------- outer loop
    dim3 go(BB / 128, DD / 128);
    dim3 gl(BB / 128, AANS / 128);
    for (int o = 0; o < OS; ++o) {
        inner_kernel<<<BB, 256, 0, stream>>>(update, g_inner, bn_inner, z, z16, latents, o);
        gemm_wmma<<<go, 256, 0, stream>>>(z16, DD, Wo1_t, DD, b_o1, h16, nullptr, DD, DD, 1);
        gemm_wmma<<<go, 256, 0, stream>>>(h16, DD, Wo2_t, DD, b_o2, nullptr, u, DD, DD, 0);
        outerz_kernel<<<BB, 256, 0, stream>>>(u, g_outer, bn_outer, g_ans, bn_ans,
                                              g_halt, bn_halt, W_halt, b_halt,
                                              z, z16, lnans, halt, o);
        gemm_wmma<<<gl, 256, 0, stream>>>(lnans, DD, Wans_t, AANS, b_ans,
                                          nullptr, logits + (size_t)o * AANS,
                                          OS * AANS, DD, 0);
    }
}
